// MultimodalBlock_88940182766108
// MI455X (gfx1250) — compile-verified, matrix operand order checked by reading, not run on hardware
//
#include <hip/hip_runtime.h>
#include <hip/hip_bf16.h>

typedef _Float16 v8h  __attribute__((ext_vector_type(8)));
typedef _Float16 v16h __attribute__((ext_vector_type(16)));
typedef float    v8f  __attribute__((ext_vector_type(8)));

#define DEV __device__ __forceinline__

// ---------------------------------------------------------------------------
// WMMA 16-bit fragment loaders. Per-lane K split (ISA 7.12.2):
// lanes 0-15 -> K {0..7, 16..23}, lanes 16-31 -> K {8..15, 24..31}.
// load_frag_ptr: p must already point at (row0+l16)*ld + k0 + hi*8.
// ---------------------------------------------------------------------------
DEV v16h load_frag_ptr(const _Float16* p) {
  v8h lo = *(const v8h*)p;
  v8h hh = *(const v8h*)(p + 16);
  v16h f;
#pragma unroll
  for (int j = 0; j < 8; ++j) { f[j] = lo[j]; f[j + 8] = hh[j]; }
  return f;
}

DEV float redmax16(float v) {
#pragma unroll
  for (int m = 1; m <= 8; m <<= 1) v = fmaxf(v, __shfl_xor(v, m, 32));
  return v;
}
DEV float redsum16(float v) {
#pragma unroll
  for (int m = 1; m <= 8; m <<= 1) v += __shfl_xor(v, m, 32);
  return v;
}
DEV float redsum32(float v) {
#pragma unroll
  for (int m = 1; m <= 16; m <<= 1) v += __shfl_xor(v, m, 32);
  return v;
}

// ---------------------------------------------------------------------------
// Pack fp32 row-major weight W[K][N] into WMMA B-fragment layout (f16).
// Fragment order matches gemm_wmma: frag = tn * tkCount + tk. blockIdx.y = batch.
// ---------------------------------------------------------------------------
__global__ void __launch_bounds__(128) pack_wfrag(
    const float* __restrict__ W, long wStride,
    _Float16* __restrict__ out, long oStride, int K, int N) {
  int lane = threadIdx.x & 31, wv = threadIdx.x >> 5;
  int l16 = lane & 15, hi = lane >> 4;
  int frag = blockIdx.x * 4 + wv;
  int z = blockIdx.y;
  int tkCount = K >> 5;
  int tk = frag % tkCount, tn = frag / tkCount;
  const float* Wb = W + (long)z * wStride;
  _Float16* ob = out + (long)z * oStride + (long)frag * 512 + lane * 16;
  int n  = tn * 16 + l16;
  int kb = tk * 32 + hi * 8;
#pragma unroll
  for (int j = 0; j < 8; ++j) ob[j]     = (_Float16)Wb[(long)(kb + j) * N + n];
#pragma unroll
  for (int j = 0; j < 8; ++j) ob[8 + j] = (_Float16)Wb[(long)(kb + 16 + j) * N + n];
}

// ---------------------------------------------------------------------------
// LayerNorm over rows of length 512; one wave per row; writes f16.
// ---------------------------------------------------------------------------
__global__ void __launch_bounds__(256) ln512(
    const float* __restrict__ x, const float* __restrict__ g,
    const float* __restrict__ bta, _Float16* __restrict__ out, int rows) {
  int lane = threadIdx.x & 31;
  int wv = threadIdx.x >> 5;
  int row = blockIdx.x * (blockDim.x >> 5) + wv;
  if (row >= rows) return;
  const float* xr = x + (long)row * 512;
  float vals[16];
  float s = 0.f;
#pragma unroll
  for (int i = 0; i < 4; ++i) {
    float4 f = *(const float4*)(xr + i * 128 + lane * 4);
    vals[i * 4 + 0] = f.x; vals[i * 4 + 1] = f.y;
    vals[i * 4 + 2] = f.z; vals[i * 4 + 3] = f.w;
    s += f.x + f.y + f.z + f.w;
  }
  s = redsum32(s);
  float mu = s * (1.f / 512.f);
  float vs = 0.f;
#pragma unroll
  for (int e = 0; e < 16; ++e) { float d = vals[e] - mu; vs += d * d; }
  vs = redsum32(vs);
  float rstd = rsqrtf(vs * (1.f / 512.f) + 1e-5f);
  _Float16* orow = out + (long)row * 512;
#pragma unroll
  for (int i = 0; i < 4; ++i)
#pragma unroll
    for (int jj = 0; jj < 4; ++jj) {
      int idx = i * 128 + lane * 4 + jj;
      orow[idx] = (_Float16)((vals[i * 4 + jj] - mu) * rstd * g[idx] + bta[idx]);
    }
}

// ---------------------------------------------------------------------------
// Generic WMMA GEMM, templated on the number of 16-col tiles per wave (NT).
// out = act(A(f16) @ Bpacked + bias) [+ res], f32 and/or f16 out.
// Wave computes 16 x (NT*16); block = 4 waves = 64 rows. Branch-free,
// software-pipelined K loop. grid covers N exactly. act: 0=none 1=tanh 2=relu.
// __launch_bounds__(128): blocks are 128 threads, so give each wave the full
// VGPR budget (no spill of the double-buffered fragments).
// ---------------------------------------------------------------------------
template <int NT>
__global__ void __launch_bounds__(128) gemm_wmma(
    const _Float16* __restrict__ A, int lda, long aStride,
    const _Float16* __restrict__ Bp, long bStride,
    const float* __restrict__ bias, int biasStride,
    const float* __restrict__ res,
    float* __restrict__ outF,
    _Float16* __restrict__ outH, long oStride,
    int N, int K, int act) {
  int wave = threadIdx.x >> 5, lane = threadIdx.x & 31;
  int l16 = lane & 15, hi = lane >> 4;
  int row0 = (blockIdx.x * 4 + wave) * 16;
  int nt0 = blockIdx.y * NT;
  int z = blockIdx.z;
  const _Float16* Ab = A + (long)z * aStride;
  const _Float16* Bb = Bp + (long)z * bStride;
  int tkCount = K >> 5;

  v8f zro = {0.f, 0.f, 0.f, 0.f, 0.f, 0.f, 0.f, 0.f};
  v8f acc[NT];
#pragma unroll
  for (int t = 0; t < NT; ++t) acc[t] = zro;

  const _Float16* aRow = Ab + (long)(row0 + l16) * lda + hi * 8;
  const _Float16* bBase = Bb + ((long)nt0 * tkCount << 9) + lane * 16;
  long bTileStride = (long)tkCount << 9;

  // prologue: K-step 0
  v16h afrag = load_frag_ptr(aRow);
  v16h bfrag[NT];
#pragma unroll
  for (int t = 0; t < NT; ++t) bfrag[t] = *(const v16h*)(bBase + t * bTileStride);

  // pipelined main loop: loads for tk+1 issued before WMMAs of tk
  for (int tk = 0; tk < tkCount - 1; ++tk) {
    const _Float16* ap = aRow + (tk + 1) * 32;
    __builtin_prefetch(ap + 32, 0, 1);  // -> global_prefetch_b8
    v16h anext = load_frag_ptr(ap);
    v16h bnext[NT];
#pragma unroll
    for (int t = 0; t < NT; ++t)
      bnext[t] = *(const v16h*)(bBase + t * bTileStride + (long)(tk + 1) * 512);
#pragma unroll
    for (int t = 0; t < NT; ++t)
      acc[t] = __builtin_amdgcn_wmma_f32_16x16x32_f16(
          false, afrag, false, bfrag[t], (short)0, acc[t], false, false);
    afrag = anext;
#pragma unroll
    for (int t = 0; t < NT; ++t) bfrag[t] = bnext[t];
  }
#pragma unroll
  for (int t = 0; t < NT; ++t)
    acc[t] = __builtin_amdgcn_wmma_f32_16x16x32_f16(
        false, afrag, false, bfrag[t], (short)0, acc[t], false, false);

  // epilogue
  const float* biasb = bias ? bias + (long)z * biasStride : nullptr;
#pragma unroll
  for (int t = 0; t < NT; ++t) {
    int col = (nt0 + t) * 16 + l16;
    float bv = biasb ? biasb[col] : 0.f;
#pragma unroll
    for (int r = 0; r < 8; ++r) {
      int row = row0 + r + hi * 8;
      float v = acc[t][r] + bv;
      if (act == 1) v = tanhf(v);
      else if (act == 2) v = fmaxf(v, 0.f);
      long idx = (long)row * N + col;
      if (res)  v += res[idx];
      if (outF) outF[(long)z * oStride + idx] = v;
      if (outH) outH[(long)z * oStride + idx] = (_Float16)v;
    }
  }
}

// ---------------------------------------------------------------------------
// Elementwise transpose: in [HB][1024 rows][ldin] (col offset) ->
// out Vt [HB][64][1024] so P@V B-fragments become contiguous b128 loads.
// ---------------------------------------------------------------------------
__global__ void __launch_bounds__(256) transpose_v(
    const _Float16* __restrict__ in, int ldin, int colOff,
    _Float16* __restrict__ out) {
  long idx = (long)blockIdx.x * blockDim.x + threadIdx.x;
  int t = (int)(idx & 1023);
  int d = (int)((idx >> 10) & 63);
  long hb = idx >> 16;
  out[idx] = in[(hb * 1024 + t) * (long)ldin + colOff + d];
}

// ---------------------------------------------------------------------------
// Causal flash attention, wave32 WMMA. Q [H][B*T][ldq], K [H][B*T][ldk],
// Vt [H*B][64][1024]. Out merged f16 [B*T][512] at column h*64.
// Wave = one 16-query strip. The 1/sqrt(HS) scale is folded into the Q
// fragments (exact power of two in f16). Strength-reduced pointers, batched
// fragment loads, causal mask only in the single diagonal tail tile.
// __launch_bounds__(128) gives each wave the full VGPR budget (no Q spills).
// ---------------------------------------------------------------------------
__global__ void __launch_bounds__(128) flash_wmma(
    const _Float16* __restrict__ Q, int ldq,
    const _Float16* __restrict__ Km, int ldk,
    const _Float16* __restrict__ Vt,
    _Float16* __restrict__ Out,
    int Bc, int Hc) {
  __shared__ __align__(32) _Float16 pLds[4][512];  // per-wave 16x32 P tile
  int wave = threadIdx.x >> 5, lane = threadIdx.x & 31;
  int l16 = lane & 15, hi = lane >> 4;
  int bh = blockIdx.y;
  int b = bh / Hc, h = bh % Hc;
  int q0 = (blockIdx.x * 4 + wave) * 16;

  const _Float16* Qb = Q  + ((long)h * Bc + b) * 1024L * ldq;
  const _Float16* Kb = Km + ((long)h * Bc + b) * 1024L * ldk;
  const _Float16* Vb = Vt + ((long)h * Bc + b) * 64L * 1024L;
  _Float16* pw = &pLds[wave][0];

  const _Float16* qPtr = Qb + (long)(q0 + l16) * ldq + hi * 8;
  v16h qf0 = load_frag_ptr(qPtr);
  v16h qf1 = load_frag_ptr(qPtr + 32);
#pragma unroll
  for (int j = 0; j < 16; ++j) {  // fold HS^-0.5 = 0.125 into Q (exact in f16)
    qf0[j] *= (_Float16)0.125f;
    qf1[j] *= (_Float16)0.125f;
  }

  v8f zro = {0.f, 0.f, 0.f, 0.f, 0.f, 0.f, 0.f, 0.f};
  v8f o[4];
#pragma unroll
  for (int t = 0; t < 4; ++t) o[t] = zro;
  float mrow[8], lrow[8];
#pragma unroll
  for (int r = 0; r < 8; ++r) { mrow[r] = -1e30f; lrow[r] = 0.f; }

  // strength-reduced per-lane pointers, bumped each key tile
  const _Float16* kPtrA = Kb + (long)l16 * ldk + hi * 8;   // n-tile 0 (keys kk..kk+15)
  const _Float16* kPtrB = kPtrA + 16L * ldk;               // n-tile 1 (keys kk+16..kk+31)
  const _Float16* vPtr  = Vb + (long)l16 * 1024 + hi * 8;  // +t*16384 per 16-dim tile
  const long kStep = 32L * ldk;
  int kk = 0;

  auto step = [&](bool masked) {
    // --- batched fragment loads (K then V); WMMAs follow, so waits are partial
    v16h kf00 = load_frag_ptr(kPtrA);        // keys kk..kk+15,  d 0..31
    v16h kf01 = load_frag_ptr(kPtrA + 32);   // keys kk..kk+15,  d 32..63
    v16h kf10 = load_frag_ptr(kPtrB);        // keys kk+16..+31, d 0..31
    v16h kf11 = load_frag_ptr(kPtrB + 32);   // keys kk+16..+31, d 32..63
    v16h vf[4];
#pragma unroll
    for (int t = 0; t < 4; ++t) vf[t] = load_frag_ptr(vPtr + t * 16384);

    v8f s0 = zro, s1 = zro;
    s0 = __builtin_amdgcn_wmma_f32_16x16x32_f16(false, qf0, false, kf00, (short)0, s0, false, false);
    s0 = __builtin_amdgcn_wmma_f32_16x16x32_f16(false, qf1, false, kf01, (short)0, s0, false, false);
    s1 = __builtin_amdgcn_wmma_f32_16x16x32_f16(false, qf0, false, kf10, (short)0, s1, false, false);
    s1 = __builtin_amdgcn_wmma_f32_16x16x32_f16(false, qf1, false, kf11, (short)0, s1, false, false);

    // --- online softmax on D-layout rows (Q already carries the scale)
    int key0 = kk + l16, key1 = kk + 16 + l16;
#pragma unroll
    for (int r = 0; r < 8; ++r) {
      int qr = q0 + r + hi * 8;
      float v0 = s0[r];
      float v1 = s1[r];
      if (masked) {
        if (key0 > qr) v0 = -1e30f;
        if (key1 > qr) v1 = -1e30f;
      }
      float mx = redmax16(fmaxf(v0, v1));
      float mnew = fmaxf(mrow[r], mx);
      float c = __expf(mrow[r] - mnew);
      float p0 = __expf(v0 - mnew);
      float p1 = __expf(v1 - mnew);
      float rs = redsum16(p0 + p1);
      lrow[r] = lrow[r] * c + rs;
      mrow[r] = mnew;
#pragma unroll
      for (int t = 0; t < 4; ++t) o[t][r] *= c;
      pw[(r + hi * 8) * 32 + l16]      = (_Float16)p0;
      pw[(r + hi * 8) * 32 + 16 + l16] = (_Float16)p1;
    }
    // --- O += P @ V  (P staged through LDS into A-fragment layout)
    v16h pf = load_frag_ptr(pw + l16 * 32 + hi * 8);
#pragma unroll
    for (int t = 0; t < 4; ++t)
      o[t] = __builtin_amdgcn_wmma_f32_16x16x32_f16(false, pf, false, vf[t], (short)0, o[t], false, false);

    kPtrA += kStep; kPtrB += kStep; vPtr += 32; kk += 32;
  };

  int nFull = (q0 + 1) >> 5;              // tiles fully below the diagonal
  for (int kt = 0; kt < nFull; ++kt) step(false);
  step(true);                             // exactly one diagonal (masked) tile

#pragma unroll
  for (int t = 0; t < 4; ++t)
#pragma unroll
    for (int r = 0; r < 8; ++r) {
      long row = (long)b * 1024 + q0 + r + hi * 8;
      float val = o[t][r] / lrow[r];
      Out[row * 512 + h * 64 + t * 16 + l16] = (_Float16)val;
    }
}

// ---------------------------------------------------------------------------
// Host orchestration
// ---------------------------------------------------------------------------
extern "C" void kernel_launch(void* const* d_in, const int* in_sizes, int n_in,
                              void* d_out, int out_size, void* d_ws, size_t ws_size,
                              hipStream_t stream) {
  (void)in_sizes; (void)n_in; (void)out_size; (void)ws_size;
  const int Bb = 8, Cc = 512, Hh = 8, BT = 8192;
  const long BTC = (long)BT * Cc;

  const float* x     = (const float*)d_in[0];
  const float* w1in[3] = {(const float*)d_in[4], (const float*)d_in[1], (const float*)d_in[7]}; // q,k,v
  const float* b1in[3] = {(const float*)d_in[5], (const float*)d_in[2], (const float*)d_in[8]};
  const float* w2in[3] = {(const float*)d_in[6], (const float*)d_in[3], (const float*)d_in[9]};
  const float* pw1 = (const float*)d_in[10]; const float* pb1 = (const float*)d_in[11];
  const float* pw2 = (const float*)d_in[12]; const float* pb2 = (const float*)d_in[13];
  const float* ffw1 = (const float*)d_in[14]; const float* ffb1 = (const float*)d_in[15];
  const float* ffw2 = (const float*)d_in[16]; const float* ffb2 = (const float*)d_in[17];
  const float* ln1g = (const float*)d_in[18]; const float* ln1b = (const float*)d_in[19];
  const float* ln2g = (const float*)d_in[20]; const float* ln2b = (const float*)d_in[21];
  const float* lncg = (const float*)d_in[22]; const float* lncb = (const float*)d_in[23];
  const float* caqw = (const float*)d_in[24]; const float* cakv = (const float*)d_in[25];
  const float* capw1 = (const float*)d_in[26]; const float* capb1 = (const float*)d_in[27];
  const float* capw2 = (const float*)d_in[28]; const float* capb2 = (const float*)d_in[29];
  float* outp = (float*)d_out;

  size_t off = 0;
  auto alloc = [&](size_t bytes) -> char* {
    char* p = (char*)d_ws + off;
    off += (bytes + 255) & ~(size_t)255;
    return p;
  };
  auto allocH = [&](size_t n) -> _Float16* { return (_Float16*)alloc(n * 2); };
  auto allocF = [&](size_t n) -> float*    { return (float*)alloc(n * 4); };
  auto fragN  = [](int K, int N, int batch) -> size_t {
    return (size_t)(K / 32) * (N / 16) * 512 * batch;
  };

  // ---- packed weights ----
  _Float16 *pk_w1[3][2], *pk_w2[3][2];
  _Float16 *pk_pw1[2], *pk_pw2[2], *pk_ffw1[2], *pk_ffw2[2];
  _Float16 *pk_caqw[2], *pk_cakv[2], *pk_capw1[2], *pk_capw2[2];
  for (int k = 0; k < 3; ++k)
    for (int i = 0; i < 2; ++i) {
      pk_w1[k][i] = allocH(fragN(512, 32, Hh));
      pk_w2[k][i] = allocH(fragN(32, 64, Hh));
    }
  for (int i = 0; i < 2; ++i) {
    pk_pw1[i]  = allocH(fragN(512, 256, 1));
    pk_pw2[i]  = allocH(fragN(256, 512, 1));
    pk_ffw1[i] = allocH(fragN(512, 2048, 1));
    pk_ffw2[i] = allocH(fragN(2048, 512, 1));
    pk_caqw[i] = allocH(fragN(512, 64, Hh));
    pk_cakv[i] = allocH(fragN(512, 128, Hh));
    pk_capw1[i] = allocH(fragN(512, 256, 1));
    pk_capw2[i] = allocH(fragN(256, 512, 1));
  }

  // ---- activations ----
  _Float16* hbuf  = allocH((size_t)BT * 512);
  _Float16* zbuf  = allocH((size_t)Hh * BT * 32);
  _Float16* qbuf  = allocH((size_t)Hh * BT * 64);
  _Float16* kbuf  = allocH((size_t)Hh * BT * 64);
  _Float16* vbuf  = allocH((size_t)Hh * BT * 64);
  _Float16* vtbuf = allocH((size_t)Hh * Bb * 64 * 1024);
  _Float16* mrg   = allocH((size_t)BT * 512);
  _Float16* t1    = allocH((size_t)BT * 256);
  _Float16* ffz   = allocH((size_t)BT * 2048);
  _Float16* kvbuf = allocH((size_t)Hh * BT * 128);
  float* x1 = allocF((size_t)BT * 512);
  float* attF[2]     = {allocF(BTC), allocF(BTC)};
  _Float16* attH[2]  = {allocH(BTC), allocH(BTC)};

  auto pack = [&](const float* W, long wStr, _Float16* op, int K, int N, int batch) {
    int frags = (K / 32) * (N / 16);
    dim3 g(frags / 4, batch);
    pack_wfrag<<<g, 128, 0, stream>>>(W, wStr, op, (long)frags * 512, K, N);
  };
  auto gemm = [&](const _Float16* A, int lda, long aStr, const _Float16* Bp, long bStr,
                  const float* bias, int biasStr, const float* res, float* oF,
                  _Float16* oH, long oStr, int N, int K, int act, int batch) {
    if (N == 32) {
      dim3 g(BT / 64, 1, batch);
      gemm_wmma<2><<<g, 128, 0, stream>>>(A, lda, aStr, Bp, bStr, bias, biasStr,
                                          res, oF, oH, oStr, N, K, act);
    } else {
      dim3 g(BT / 64, N / 64, batch);
      gemm_wmma<4><<<g, 128, 0, stream>>>(A, lda, aStr, Bp, bStr, bias, biasStr,
                                          res, oF, oH, oStr, N, K, act);
    }
  };

  // ---- pack everything ----
  for (int k = 0; k < 3; ++k)
    for (int i = 0; i < 2; ++i) {
      pack(w1in[k] + (long)i * Hh * 512 * 32, (long)512 * 32, pk_w1[k][i], 512, 32, Hh);
      pack(w2in[k] + (long)i * Hh * 32 * 64,  (long)32 * 64,  pk_w2[k][i], 32, 64, Hh);
    }
  for (int i = 0; i < 2; ++i) {
    pack(pw1  + (long)i * 512 * 256,  0, pk_pw1[i],  512, 256, 1);
    pack(pw2  + (long)i * 256 * 512,  0, pk_pw2[i],  256, 512, 1);
    pack(ffw1 + (long)i * 512 * 2048, 0, pk_ffw1[i], 512, 2048, 1);
    pack(ffw2 + (long)i * 2048 * 512, 0, pk_ffw2[i], 2048, 512, 1);
    pack(caqw + (long)i * Hh * 512 * 64,  (long)512 * 64,  pk_caqw[i], 512, 64, Hh);
    pack(cakv + (long)i * Hh * 512 * 128, (long)512 * 128, pk_cakv[i], 512, 128, Hh);
    pack(capw1 + (long)i * 512 * 256, 0, pk_capw1[i], 512, 256, 1);
    pack(capw2 + (long)i * 256 * 512, 0, pk_capw2[i], 256, 512, 1);
  }

  // ---- self-attention + FF per modality ----
  for (int i = 0; i < 2; ++i) {
    const float* xi = x + (long)i * BTC;
    ln512<<<BT / 8, 256, 0, stream>>>(xi, ln1g + i * 512, ln1b + i * 512, hbuf, BT);
    for (int k = 0; k < 3; ++k) {
      _Float16* dst = (k == 0) ? qbuf : (k == 1) ? kbuf : vbuf;
      gemm(hbuf, 512, 0, pk_w1[k][i], (long)32 * 512, b1in[k] + (long)i * Hh * 32, 32,
           nullptr, nullptr, zbuf, (long)BT * 32, 32, 512, 1, Hh);
      gemm(zbuf, 32, (long)BT * 32, pk_w2[k][i], (long)4 * 512, nullptr, 0,
           nullptr, nullptr, dst, (long)BT * 64, 64, 32, 0, Hh);
    }
    transpose_v<<<16384, 256, 0, stream>>>(vbuf, 64, 0, vtbuf);
    flash_wmma<<<dim3(16, Bb * Hh), 128, 0, stream>>>(qbuf, 64, kbuf, 64, vtbuf, mrg, Bb, Hh);
    gemm(mrg, 512, 0, pk_pw1[i], 0, pb1 + i * 256, 0, nullptr, nullptr, t1, 0, 256, 512, 1, 1);
    gemm(t1, 256, 0, pk_pw2[i], 0, pb2 + i * 512, 0, xi, x1, nullptr, 0, 512, 256, 0, 1);
    ln512<<<BT / 8, 256, 0, stream>>>(x1, ln2g + i * 512, ln2b + i * 512, hbuf, BT);
    gemm(hbuf, 512, 0, pk_ffw1[i], 0, ffb1 + i * 2048, 0, nullptr, nullptr, ffz, 0, 2048, 512, 2, 1);
    gemm(ffz, 2048, 0, pk_ffw2[i], 0, ffb2 + i * 512, 0, x1, attF[i], attH[i], 0, 512, 2048, 0, 1);
  }

  // ---- cross-attention per modality ----
  for (int i = 0; i < 2; ++i) {
    int j = 1 - i;
    ln512<<<BT / 8, 256, 0, stream>>>(attF[i], lncg + i * 512, lncb + i * 512, hbuf, BT);
    gemm(hbuf, 512, 0, pk_caqw[i], (long)64 * 512, nullptr, 0,
         nullptr, nullptr, qbuf, (long)BT * 64, 64, 512, 0, Hh);
    gemm(attH[j], 512, 0, pk_cakv[i], (long)128 * 512, nullptr, 0,
         nullptr, nullptr, kvbuf, (long)BT * 128, 128, 512, 0, Hh);
    transpose_v<<<16384, 256, 0, stream>>>(kvbuf, 128, 64, vtbuf);
    flash_wmma<<<dim3(16, Bb * Hh), 128, 0, stream>>>(qbuf, 64, kvbuf, 128, vtbuf, mrg, Bb, Hh);
    gemm(mrg, 512, 0, pk_capw1[i], 0, capb1 + i * 256, 0, nullptr, nullptr, t1, 0, 256, 512, 1, 1);
    gemm(t1, 256, 0, pk_capw2[i], 0, capb2 + i * 512, 0, attF[i], outp + (long)i * BTC,
         nullptr, 0, 512, 256, 0, 1);
  }
}
